// MambaBlock_7241314861837
// MI455X (gfx1250) — matrix operand order repaired and checked by reading, not compiled, not run on hardware
//
#include <hip/hip_runtime.h>
#include <hip/hip_bf16.h>
#include <stddef.h>

// ---------------- problem constants ----------------
#define D_MODEL   768
#define D_INNER   1536
#define D_STATE   16
#define D_CONV    4
#define BATCH     2
#define SEQ       2048
#define TOKENS    (BATCH * SEQ)          // 4096
#define N_XZ      (2 * D_INNER)          // 3072
#define N_DBC     (2 * D_STATE + 1)      // 33
#define CHUNKS    16
#define CHUNK_LEN (SEQ / CHUNKS)         // 128

// ---------------- workspace layout (bytes, all 256B aligned) ----------------
static const size_t SCALES_OFF = 0;                                   // 256 B
static const size_t XN_OFF     = 256;                                 // f16 4096x768
static const size_t WQIN_OFF   = XN_OFF     + (size_t)TOKENS*D_MODEL*2;        // f16 3072x768
static const size_t WQOUT_OFF  = WQIN_OFF   + (size_t)N_XZ*D_MODEL*2;          // f16 768x1536
static const size_t XZ_OFF     = WQOUT_OFF  + (size_t)D_MODEL*D_INNER*2;       // f32 4096x3072
static const size_t XC_OFF     = XZ_OFF     + (size_t)TOKENS*N_XZ*4;           // f32 4096x1536
static const size_t XDBC_OFF   = XC_OFF     + (size_t)TOKENS*D_INNER*4;        // f32 4096x33
static const size_t YRMS_OFF   = XDBC_OFF   + (size_t)TOKENS*N_DBC*4;          // f16 4096x1536
static const size_t CARRYP_OFF = YRMS_OFF   + (size_t)TOKENS*D_INNER*2;        // f32 2*16*1536*16
static const size_t CARRYS_OFF = CARRYP_OFF + (size_t)BATCH*CHUNKS*D_INNER*D_STATE*4;

// ---------------- vector types for WMMA ----------------
typedef __attribute__((ext_vector_type(16))) _Float16 v16h;
typedef __attribute__((ext_vector_type(8)))  _Float16 v8h;
typedef __attribute__((ext_vector_type(8)))  float    v8f;

// WGP-scope (scope 0) speculative prefetch: fills all cache levels incl. the
// WGP cache on miss (ISA 10.5). __builtin_prefetch can only reach SE/SYS
// scope, which skips the WGP cache, so use inline asm with the default scope.
__device__ __forceinline__ void prefetch_wgp(const void* p) {
    asm volatile("global_prefetch_b8 %0, off" :: "v"(p));
}

// ======================================================================
// mean(|w|) over a whole matrix (single block -> deterministic)
// ======================================================================
__global__ __launch_bounds__(256) void absmean_kernel(const float* __restrict__ w,
                                                      int n, float* __restrict__ out) {
    __shared__ float red[256];
    float s = 0.f;
    for (int i = threadIdx.x; i < n; i += 256) s += fabsf(w[i]);
    red[threadIdx.x] = s;
    __syncthreads();
    for (int off = 128; off > 0; off >>= 1) {
        if ((int)threadIdx.x < off) red[threadIdx.x] += red[threadIdx.x + off];
        __syncthreads();
    }
    if (threadIdx.x == 0) out[0] = fmaxf(red[0] / (float)n, 1e-5f);
}

// ======================================================================
// ternary quantize: wq = round(clip(w/scale, -1, 1)) as f16 (RNE, like jnp.round)
// ======================================================================
__global__ __launch_bounds__(256) void quantize_kernel(const float* __restrict__ w,
                                                       _Float16* __restrict__ wq,
                                                       int n, const float* __restrict__ scale) {
    int i = blockIdx.x * 256 + threadIdx.x;
    if (i >= n) return;
    float v = w[i] / scale[0];
    v = fminf(fmaxf(v, -1.f), 1.f);
    wq[i] = (_Float16)rintf(v);
}

// ======================================================================
// rmsnorm over last dim D (row stride in_stride), output f16 rows of length D
// ======================================================================
__global__ __launch_bounds__(256) void rmsnorm_kernel(const float* __restrict__ x,
                                                      const float* __restrict__ w,
                                                      _Float16* __restrict__ out,
                                                      int D, int in_stride) {
    int row = blockIdx.x;
    const float* xr = x + (size_t)row * in_stride;
    __shared__ float red[256];
    __shared__ float norm;
    float s = 0.f;
    for (int i = threadIdx.x; i < D; i += 256) { float v = xr[i]; s += v * v; }
    red[threadIdx.x] = s;
    __syncthreads();
    for (int off = 128; off > 0; off >>= 1) {
        if ((int)threadIdx.x < off) red[threadIdx.x] += red[threadIdx.x + off];
        __syncthreads();
    }
    if (threadIdx.x == 0) norm = rsqrtf(red[0] / (float)D + 1e-5f);
    __syncthreads();
    float nm = norm;
    for (int i = threadIdx.x; i < D; i += 256)
        out[(size_t)row * D + i] = (_Float16)(xr[i] * nm * w[i]);
}

// ======================================================================
// WMMA fragment helpers (A/B use contiguous k, same in both, not ISA table order; C/D per ISA 7.12.2)
//   A 16x32:  lane L (m=L%16): halves j   -> k = 16*(L/16)+j (contiguous in A row m)
//   B 32x16:  lane L (n=L%16): halves j   -> k = 16*(L/16)+j (contiguous in Wq row n)
//   C/D:      lane L (n=L%16): VGPR v -> m = v + 8*(L/16)
// ======================================================================
__device__ __forceinline__ v16h load_afrag(const _Float16* __restrict__ arow,
                                           int k, int a_off) {
    v8h lo = *(const v8h*)(arow + k + a_off);
    v8h hi = *(const v8h*)(arow + k + a_off + 8);
    v16h f;
#pragma unroll
    for (int j = 0; j < 8; ++j) { f[j] = lo[j]; f[j + 8] = hi[j]; }
    return f;
}

__device__ __forceinline__ void store_tile(float* __restrict__ C,
                                           const float* __restrict__ residual,
                                           int N, int row0, int col,
                                           v8f acc, float sc, int half_id) {
#pragma unroll
    for (int v = 0; v < 8; ++v) {
        int row = row0 + v + 8 * half_id;
        size_t idx = (size_t)row * N + col;
        float val = acc[v] * sc;
        if (residual) val += residual[idx];
        C[idx] = val;
    }
}

// ======================================================================
// WMMA GEMM: C[M,N] = (A[M,K] @ Wq[N,K]^T) * scale (+ residual)
// 2x2 register tile per wave (32x32 output): 4 fragments -> 4 WMMAs per
// 32-deep k-step = 16 flop/byte from cache. Manually unrolled 2x over k
// (volatile asm inhibits #pragma unroll): 8 independent WMMAs per iteration.
// One set of WGP-scope prefetches per 64-k iteration: each lane's row
// advances 64 B per k-step, so one 128 B line covers both steps.
// ======================================================================
__global__ __launch_bounds__(256) void gemm_wmma_kernel(const _Float16* __restrict__ A,
                                                        const _Float16* __restrict__ Wq,
                                                        float* __restrict__ C,
                                                        const float* __restrict__ scale_p,
                                                        const float* __restrict__ residual,
                                                        int M, int N, int K) {
    const int wave = threadIdx.x >> 5;
    const int lane = threadIdx.x & 31;
    const int btiles_n = N >> 5;                 // 32-wide output blocks
    const int bt  = blockIdx.x * 8 + wave;
    const int btm = bt / btiles_n;
    const int btn = bt % btiles_n;
    if (btm >= (M >> 5)) return;

    const int half_id = lane >> 4;   // 0 or 1
    const int lrow    = lane & 15;
    const _Float16* __restrict__ arow0 = A  + (size_t)(btm * 32 + lrow) * K;
    const _Float16* __restrict__ arow1 = arow0 + (size_t)16 * K;
    const _Float16* __restrict__ brow0 = Wq + (size_t)(btn * 32 + lrow) * K;
    const _Float16* __restrict__ brow1 = brow0 + (size_t)16 * K;
    const int a_off = half_id * 16;
    const int b_off = half_id * 16;

    v8f acc00 = {}, acc01 = {}, acc10 = {}, acc11 = {};
    for (int k = 0; k < K; k += 64) {
        // WGP-scope speculative prefetch of the next 64-k iteration's lines
        // (global_prefetch_b8, scope 0; silently dropped if OOB)
        prefetch_wgp((const void*)(arow0 + k + 64 + a_off));
        prefetch_wgp((const void*)(arow1 + k + 64 + a_off));
        prefetch_wgp((const void*)(brow0 + k + 64 + b_off));
        prefetch_wgp((const void*)(brow1 + k + 64 + b_off));

        // ---- k-step 0 ----
        {
            v16h a0 = load_afrag(arow0, k, a_off);
            v16h a1 = load_afrag(arow1, k, a_off);
            v16h b0 = *(const v16h*)(brow0 + k + b_off);
            v16h b1 = *(const v16h*)(brow1 + k + b_off);
            acc00 = __builtin_amdgcn_wmma_f32_16x16x32_f16(false, a0, false, b0,
                                                           (short)0, acc00, false, false);
            acc01 = __builtin_amdgcn_wmma_f32_16x16x32_f16(false, a0, false, b1,
                                                           (short)0, acc01, false, false);
            acc10 = __builtin_amdgcn_wmma_f32_16x16x32_f16(false, a1, false, b0,
                                                           (short)0, acc10, false, false);
            acc11 = __builtin_amdgcn_wmma_f32_16x16x32_f16(false, a1, false, b1,
                                                           (short)0, acc11, false, false);
        }
        // ---- k-step 1 ----
        {
            v16h a0 = load_afrag(arow0, k + 32, a_off);
            v16h a1 = load_afrag(arow1, k + 32, a_off);
            v16h b0 = *(const v16h*)(brow0 + k + 32 + b_off);
            v16h b1 = *(const v16h*)(brow1 + k + 32 + b_off);
            acc00 = __builtin_amdgcn_wmma_f32_16x16x32_f16(false, a0, false, b0,
                                                           (short)0, acc00, false, false);
            acc01 = __builtin_amdgcn_wmma_f32_16x16x32_f16(false, a0, false, b1,
                                                           (short)0, acc01, false, false);
            acc10 = __builtin_amdgcn_wmma_f32_16x16x32_f16(false, a1, false, b0,
                                                           (short)0, acc10, false, false);
            acc11 = __builtin_amdgcn_wmma_f32_16x16x32_f16(false, a1, false, b1,
                                                           (short)0, acc11, false, false);
        }
    }

    const float sc = scale_p[0];
    const int r0 = btm * 32;
    const int c0 = btn * 32 + lrow;
    store_tile(C, residual, N, r0,      c0,      acc00, sc, half_id);
    store_tile(C, residual, N, r0,      c0 + 16, acc01, sc, half_id);
    store_tile(C, residual, N, r0 + 16, c0,      acc10, sc, half_id);
    store_tile(C, residual, N, r0 + 16, c0 + 16, acc11, sc, half_id);
}

// ======================================================================
// depthwise causal conv (k=4) + bias + SiLU; reads x_path half of xz
// ======================================================================
__global__ __launch_bounds__(256) void conv_silu_kernel(const float* __restrict__ xz,
                                                        const float* __restrict__ cw,
                                                        const float* __restrict__ cb,
                                                        float* __restrict__ xc) {
    int idx = blockIdx.x * 256 + threadIdx.x;
    if (idx >= TOKENS * D_INNER) return;
    int d = idx % D_INNER;
    int l = (idx / D_INNER) % SEQ;
    int b = idx / (D_INNER * SEQ);
    float acc = cb[d];
#pragma unroll
    for (int k = 0; k < D_CONV; ++k) {
        int li = l - (D_CONV - 1) + k;
        if (li >= 0)
            acc += cw[d * D_CONV + k] * xz[((size_t)b * SEQ + li) * N_XZ + d];
    }
    float sig = 1.f / (1.f + __expf(-acc));
    xc[idx] = acc * sig;
}

// ======================================================================
// x_dbc[row, 0..32] = xc_row @ x_proj_weight^T   (33 outputs, K=1536)
// ======================================================================
__global__ __launch_bounds__(256) void xproj_kernel(const float* __restrict__ xc,
                                                    const float* __restrict__ xpw,
                                                    float* __restrict__ x_dbc) {
    __shared__ float row[D_INNER];
    int r = blockIdx.x;
    const float* src = xc + (size_t)r * D_INNER;
    for (int i = threadIdx.x; i < D_INNER; i += 256) row[i] = src[i];
    __syncthreads();
    if (threadIdx.x < N_DBC) {
        const float* wr = xpw + threadIdx.x * D_INNER;
        float acc = 0.f;
        for (int i = 0; i < D_INNER; ++i) acc += row[i] * wr[i];
        x_dbc[(size_t)r * N_DBC + threadIdx.x] = acc;
    }
}

// ======================================================================
// chunked selective scan — pass 1: per-chunk Pi(dA) and zero-init state
// block = (batch, d-block of 256, chunk); thread owns one channel d, 16 states
// ======================================================================
__global__ __launch_bounds__(256) void scan_pass1_kernel(const float* __restrict__ x_dbc,
                                                         const float* __restrict__ xc,
                                                         const float* __restrict__ A_log,
                                                         const float* __restrict__ dtw,
                                                         const float* __restrict__ dtb,
                                                         float* __restrict__ carryP,
                                                         float* __restrict__ carryS) {
    const int blk  = blockIdx.x;             // b*(6*16) + dblk*16 + c
    const int b    = blk / (6 * CHUNKS);
    const int rem  = blk % (6 * CHUNKS);
    const int dblk = rem / CHUNKS;
    const int c    = rem % CHUNKS;
    const int d    = dblk * 256 + threadIdx.x;

    float A[D_STATE], h[D_STATE], P[D_STATE];
#pragma unroll
    for (int n = 0; n < D_STATE; ++n) {
        float al = A_log[d * D_STATE + n];
        al = fminf(fmaxf(al, -20.f), 2.f);
        A[n] = -__expf(al);
        h[n] = 0.f;
        P[n] = 1.f;
    }
    const float w = dtw[d], bia = dtb[d];
    __shared__ float bc[1 + D_STATE];

    const int t0 = c * CHUNK_LEN;
    for (int t = t0; t < t0 + CHUNK_LEN; ++t) {
        size_t tok = (size_t)b * SEQ + t;
        if (threadIdx.x < 1 + D_STATE) bc[threadIdx.x] = x_dbc[tok * N_DBC + threadIdx.x];
        __syncthreads();
        float v = fmaf(bc[0], w, bia);
        float delta = (v > 20.f) ? v : log1pf(__expf(v));
        float du = delta * xc[tok * D_INNER + d];
#pragma unroll
        for (int n = 0; n < D_STATE; ++n) {
            float dA = __expf(delta * A[n]);
            h[n] = fmaf(dA, h[n], du * bc[1 + n]);
            P[n] *= dA;
        }
        __syncthreads();
    }
    size_t base = (((size_t)b * CHUNKS + c) * D_INNER + d) * D_STATE;
#pragma unroll
    for (int n = 0; n < D_STATE; ++n) { carryP[base + n] = P[n]; carryS[base + n] = h[n]; }
}

// ======================================================================
// combine carries across chunks; rewrite carryS[c] as the INIT state of chunk c
// ======================================================================
__global__ __launch_bounds__(256) void scan_combine_kernel(const float* __restrict__ carryP,
                                                           float* __restrict__ carryS) {
    int idx = blockIdx.x * 256 + threadIdx.x;       // over BATCH*D_INNER*D_STATE
    if (idx >= BATCH * D_INNER * D_STATE) return;
    int b  = idx / (D_INNER * D_STATE);
    int dn = idx % (D_INNER * D_STATE);
    float H = 0.f;
    for (int c = 0; c < CHUNKS; ++c) {
        size_t off = ((size_t)b * CHUNKS + c) * (D_INNER * D_STATE) + dn;
        float p = carryP[off];
        float s = carryS[off];
        carryS[off] = H;          // init state for this chunk
        H = fmaf(p, H, s);
    }
}

// ======================================================================
// pass 2: replay chunk from its carry; fuse y = sum(h*C) + x*D, * silu(z);
// write y over the dead x_path half of xz (in place, layout b,l,d)
// ======================================================================
__global__ __launch_bounds__(256) void scan_pass2_kernel(const float* __restrict__ x_dbc,
                                                         const float* __restrict__ xc,
                                                         const float* __restrict__ A_log,
                                                         const float* __restrict__ dtw,
                                                         const float* __restrict__ dtb,
                                                         const float* __restrict__ Dp,
                                                         const float* __restrict__ init,
                                                         float* __restrict__ xz) {
    const int blk  = blockIdx.x;
    const int b    = blk / (6 * CHUNKS);
    const int rem  = blk % (6 * CHUNKS);
    const int dblk = rem / CHUNKS;
    const int c    = rem % CHUNKS;
    const int d    = dblk * 256 + threadIdx.x;

    float A[D_STATE], h[D_STATE];
    size_t base = (((size_t)b * CHUNKS + c) * D_INNER + d) * D_STATE;
#pragma unroll
    for (int n = 0; n < D_STATE; ++n) {
        float al = A_log[d * D_STATE + n];
        al = fminf(fmaxf(al, -20.f), 2.f);
        A[n] = -__expf(al);
        h[n] = init[base + n];
    }
    const float w = dtw[d], bia = dtb[d], Dd = Dp[d];
    __shared__ float bc[N_DBC];

    const int t0 = c * CHUNK_LEN;
    for (int t = t0; t < t0 + CHUNK_LEN; ++t) {
        size_t tok = (size_t)b * SEQ + t;
        if (threadIdx.x < N_DBC) bc[threadIdx.x] = x_dbc[tok * N_DBC + threadIdx.x];
        __syncthreads();
        float v = fmaf(bc[0], w, bia);
        float delta = (v > 20.f) ? v : log1pf(__expf(v));
        float xval = xc[tok * D_INNER + d];
        float du = delta * xval;
        float y = 0.f;
#pragma unroll
        for (int n = 0; n < D_STATE; ++n) {
            float dA = __expf(delta * A[n]);
            h[n] = fmaf(dA, h[n], du * bc[1 + n]);
            y = fmaf(h[n], bc[1 + D_STATE + n], y);
        }
        float zv  = xz[tok * N_XZ + D_INNER + d];
        float sig = 1.f / (1.f + __expf(-zv));
        float out = (y + xval * Dd) * (zv * sig);
        xz[tok * N_XZ + d] = out;
        __syncthreads();
    }
}

// ======================================================================
// host side
// ======================================================================
extern "C" void kernel_launch(void* const* d_in, const int* in_sizes, int n_in,
                              void* d_out, int out_size, void* d_ws, size_t ws_size,
                              hipStream_t stream) {
    (void)in_sizes; (void)n_in; (void)out_size; (void)ws_size;
    const float* x        = (const float*)d_in[0];
    const float* norm_w   = (const float*)d_in[1];
    const float* in_proj  = (const float*)d_in[2];
    const float* conv_w   = (const float*)d_in[3];
    const float* conv_b   = (const float*)d_in[4];
    const float* xproj_w  = (const float*)d_in[5];
    const float* dt_w     = (const float*)d_in[6];
    const float* dt_b     = (const float*)d_in[7];
    const float* A_log    = (const float*)d_in[8];
    const float* Dp       = (const float*)d_in[9];
    const float* onorm_w  = (const float*)d_in[10];
    const float* out_proj = (const float*)d_in[11];
    float* out = (float*)d_out;

    char* ws = (char*)d_ws;
    float*    scales = (float*)(ws + SCALES_OFF);
    _Float16* xn     = (_Float16*)(ws + XN_OFF);
    _Float16* wq_in  = (_Float16*)(ws + WQIN_OFF);
    _Float16* wq_out = (_Float16*)(ws + WQOUT_OFF);
    float*    xz     = (float*)(ws + XZ_OFF);
    float*    xc     = (float*)(ws + XC_OFF);
    float*    x_dbc  = (float*)(ws + XDBC_OFF);
    _Float16* yrms   = (_Float16*)(ws + YRMS_OFF);
    float*    carryP = (float*)(ws + CARRYP_OFF);
    float*    carryS = (float*)(ws + CARRYS_OFF);

    const int n_in_w  = N_XZ * D_MODEL;       // 3072*768
    const int n_out_w = D_MODEL * D_INNER;    // 768*1536

    // 1) weight scales (mean |w|)
    absmean_kernel<<<1, 256, 0, stream>>>(in_proj,  n_in_w,  scales + 0);
    absmean_kernel<<<1, 256, 0, stream>>>(out_proj, n_out_w, scales + 1);

    // 2) ternary-quantize weights to f16
    quantize_kernel<<<(n_in_w  + 255) / 256, 256, 0, stream>>>(in_proj,  wq_in,  n_in_w,  scales + 0);
    quantize_kernel<<<(n_out_w + 255) / 256, 256, 0, stream>>>(out_proj, wq_out, n_out_w, scales + 1);

    // 3) rmsnorm(x) -> f16
    rmsnorm_kernel<<<TOKENS, 256, 0, stream>>>(x, norm_w, xn, D_MODEL, D_MODEL);

    // 4) GEMM1: xz = (xn @ wq_in^T) * scale_in   [4096 x 3072]
    {
        int wave_tiles = (TOKENS / 32) * (N_XZ / 32);     // 128*96 = 12288
        gemm_wmma_kernel<<<wave_tiles / 8, 256, 0, stream>>>(xn, wq_in, xz, scales + 0,
                                                             nullptr, TOKENS, N_XZ, D_MODEL);
    }

    // 5) depthwise conv + SiLU -> xc
    conv_silu_kernel<<<(TOKENS * D_INNER + 255) / 256, 256, 0, stream>>>(xz, conv_w, conv_b, xc);

    // 6) x_dbc = xc @ x_proj_weight^T   [4096 x 33]
    xproj_kernel<<<TOKENS, 256, 0, stream>>>(xc, xproj_w, x_dbc);

    // 7) chunked selective scan (pass1 -> combine -> pass2, y written into xz)
    {
        int nblk = BATCH * (D_INNER / 256) * CHUNKS;   // 2*6*16 = 192
        scan_pass1_kernel<<<nblk, 256, 0, stream>>>(x_dbc, xc, A_log, dt_w, dt_b, carryP, carryS);
        scan_combine_kernel<<<(BATCH * D_INNER * D_STATE + 255) / 256, 256, 0, stream>>>(carryP, carryS);
        scan_pass2_kernel<<<nblk, 256, 0, stream>>>(x_dbc, xc, A_log, dt_w, dt_b, Dp, carryS, xz);
    }

    // 8) rmsnorm(y) -> f16 (y rows live in xz with stride 3072)
    rmsnorm_kernel<<<TOKENS, 256, 0, stream>>>(xz, onorm_w, yrms, D_INNER, N_XZ);

    // 9) GEMM2: out = (yrms @ wq_out^T) * scale_out + x   [4096 x 768]
    {
        int wave_tiles = (TOKENS / 32) * (D_MODEL / 32);  // 128*24 = 3072
        gemm_wmma_kernel<<<wave_tiles / 8, 256, 0, stream>>>(yrms, wq_out, out, scales + 1,
                                                             x, TOKENS, D_MODEL, D_INNER);
    }
}
